// CNF__11115375362835
// MI455X (gfx1250) — matrix operand order repaired and checked by reading, not compiled, not run on hardware
//
#include <hip/hip_runtime.h>

// CNF forward + exact Jacobian trace, fused per 2-batch-row workgroup.
// Tangent propagation T_{k+1} = diag(1-h^2) * (W_k @ T_k) and the forward
// hidden layer share the left matrix W_k -> one WMMA GEMM with N = 2*32 + 2.

#define HDIM 1024
#define DDIM 32
#define BATCH 4096
#define NB 2                      // batch rows per workgroup
#define TSTRIDE 1032              // bf16 elems per Tbuf column (1024 + 8 pad -> conflict-free)
#define NCOLS 80                  // 64 tangent cols + 2 h cols, padded to 5 N-tiles
#define RED_OFF (NCOLS * TSTRIDE) // ushort offset where f32 reduction area begins
#define LDS_BYTES (RED_OFF * 2 + 2112 * 4)

typedef __bf16 bf16x16 __attribute__((ext_vector_type(16)));
typedef float f32x8 __attribute__((ext_vector_type(8)));
typedef unsigned short u16x8 __attribute__((ext_vector_type(8)));
typedef unsigned short u16x16 __attribute__((ext_vector_type(16)));

__device__ __forceinline__ unsigned short f2bf(float f) {
  unsigned u = __builtin_bit_cast(unsigned, f);
  unsigned r = u + 0x7FFFu + ((u >> 16) & 1u);  // round-to-nearest-even
  return (unsigned short)(r >> 16);
}
__device__ __forceinline__ float bf2f(unsigned short h) {
  unsigned u = ((unsigned)h) << 16;
  return __builtin_bit_cast(float, u);
}
__device__ __forceinline__ bf16x16 mk16(u16x8 lo, u16x8 hi) {
  u16x16 v = __builtin_shufflevector(lo, hi, 0, 1, 2, 3, 4, 5, 6, 7,
                                     8, 9, 10, 11, 12, 13, 14, 15);
  return __builtin_bit_cast(bf16x16, v);
}

// One hidden layer: C = W @ [T | h]; h' = tanh(C_h + bias); T' = (1-h'^2) * C_T.
// Tbuf updated in place (barrier-separated phases). Wave w owns M rows [64w,64w+64).
__device__ __forceinline__ void hidden_layer(unsigned short* Tbuf,
                                             const unsigned short* __restrict__ Wbf,
                                             const float* __restrict__ bias,
                                             int wave, int lane) {
  const int l15 = lane & 15;
  const int hi = (lane >> 4) & 1;

  f32x8 C[4][5] = {};

  for (int k0 = 0; k0 < HDIM; k0 += 32) {
    bf16x16 Bt[5];
#pragma unroll
    for (int nt = 0; nt < 5; ++nt) {
      const int col = nt * 16 + l15;
      const u16x8* bp = (const u16x8*)&Tbuf[col * TSTRIDE + k0 + hi * 16];
      Bt[nt] = mk16(bp[0], bp[1]);  // 2x ds_load_b128, contiguous k = 16h + s (matches A)
    }
#pragma unroll
    for (int mt = 0; mt < 4; ++mt) {
      const int row = wave * 64 + mt * 16 + l15;
      const u16x8* ap = (const u16x8*)&Wbf[(size_t)row * HDIM + k0 + hi * 16];
      bf16x16 A = mk16(ap[0], ap[1]);  // 2x global_load_b128, contiguous k = 16h + s (matches B)
#pragma unroll
      for (int nt = 0; nt < 5; ++nt)
        C[mt][nt] = __builtin_amdgcn_wmma_f32_16x16x32_bf16(
            false, A, false, Bt[nt], (short)0, C[mt][nt], false, false);
    }
  }
  __syncthreads();  // everyone done reading Tbuf

  // h columns (cols 64,65): bias + tanh, store new h (bf16) back to Tbuf.
  if (l15 < NB) {
    const int b = l15;
#pragma unroll
    for (int mt = 0; mt < 4; ++mt) {
      const int a0 = wave * 64 + mt * 16 + hi * 8;
      u16x8 hv;
#pragma unroll
      for (int r = 0; r < 8; ++r) {
        float hval = tanhf(C[mt][4][r] + bias[a0 + r]);
        hv[r] = f2bf(hval);
      }
      *(u16x8*)&Tbuf[(64 + b) * TSTRIDE + a0] = hv;
    }
  }
  __syncthreads();  // new h visible

  // Tangent columns: scale by s = 1 - h'^2 (per output row), store bf16.
#pragma unroll
  for (int mt = 0; mt < 4; ++mt) {
    const int a0 = wave * 64 + mt * 16 + hi * 8;
#pragma unroll
    for (int nt = 0; nt < 4; ++nt) {
      const int col = nt * 16 + l15;
      const int b = col >> 5;
      u16x8 hv = *(const u16x8*)&Tbuf[(64 + b) * TSTRIDE + a0];  // broadcast read
      u16x8 tv;
#pragma unroll
      for (int r = 0; r < 8; ++r) {
        float hval = bf2f(hv[r]);
        tv[r] = f2bf((1.0f - hval * hval) * C[mt][nt][r]);
      }
      *(u16x8*)&Tbuf[col * TSTRIDE + a0] = tv;
    }
  }
  __syncthreads();
}

__global__ void __launch_bounds__(512, 1)
cnf_fused(const float* __restrict__ t, const float* __restrict__ z,
          const float* __restrict__ W0, const float* __restrict__ b0,
          const unsigned short* __restrict__ W1bf, const float* __restrict__ b1,
          const unsigned short* __restrict__ W2bf, const float* __restrict__ b2,
          const float* __restrict__ W3, const float* __restrict__ b3,
          float* __restrict__ out) {
  extern __shared__ unsigned short smem[];
  unsigned short* Tbuf = smem;
  float* red = (float*)&smem[RED_OFF];  // 2112 floats

  const int tid = threadIdx.x;
  const int lane = tid & 31;
  const int wave = tid >> 5;
  const int bg = blockIdx.x * NB;

  // ---------- layer 0: h1 = tanh(W0 @ [z;t] + b0), T1 = diag(1-h1^2) @ W0[:, :32]
  {
    const float tval = t[0];
    for (int rr = 0; rr < 2; ++rr) {
      const int a = tid + rr * 512;
      const float* w0r = W0 + (size_t)a * (DDIM + 1);
      float wz[DDIM + 1];
      for (int j = 0; j <= DDIM; ++j) wz[j] = w0r[j];
      for (int b = 0; b < NB; ++b) {
        const float* zb = z + (size_t)(bg + b) * DDIM;
        float pre = wz[DDIM] * tval + b0[a];
        for (int j = 0; j < DDIM; ++j) pre += wz[j] * zb[j];
        float hval = tanhf(pre);
        Tbuf[(64 + b) * TSTRIDE + a] = f2bf(hval);
        float s = 1.0f - hval * hval;
        for (int n = 0; n < DDIM; ++n)
          Tbuf[(b * DDIM + n) * TSTRIDE + a] = f2bf(s * wz[n]);
      }
    }
  }
  __syncthreads();

  // ---------- hidden layers (fused forward + tangent GEMMs on WMMA)
  hidden_layer(Tbuf, W1bf, b1, wave, lane);
  hidden_layer(Tbuf, W2bf, b2, wave, lane);

  // ---------- final layer: out = W3 @ h3 + b3 ; trace_b = sum_i (W3 @ T3)[i, i]
  // Deterministic tree reduction (no FP atomics).
  float* part_out = red;         // [b][i][g] : 2*32*16
  float* part_tr = red + 1024;   // [b][i][g]
  float* tr_i = red + 2048;      // [b][i]    : 64

  const int i = tid & 31;
  const int g = tid >> 5;  // 16 K-chunks of 64
  const int kb = g * 64;
  const float* w3r = W3 + (size_t)i * HDIM;
  for (int b = 0; b < NB; ++b) {
    const unsigned short* tc = &Tbuf[(b * DDIM + i) * TSTRIDE];
    const unsigned short* hc = &Tbuf[(64 + b) * TSTRIDE];
    float tr = 0.0f, ot = 0.0f;
    for (int k = kb; k < kb + 64; ++k) {
      float w = w3r[k];
      tr += w * bf2f(tc[k]);
      ot += w * bf2f(hc[k]);
    }
    part_out[(b * DDIM + i) * 16 + g] = ot;
    part_tr[(b * DDIM + i) * 16 + g] = tr;
  }
  __syncthreads();
  if (tid < 64) {
    const int b = tid >> 5, ii = tid & 31;
    float so = 0.0f, st = 0.0f;
    for (int g2 = 0; g2 < 16; ++g2) {
      so += part_out[(b * DDIM + ii) * 16 + g2];
      st += part_tr[(b * DDIM + ii) * 16 + g2];
    }
    out[(size_t)(bg + b) * DDIM + ii] = so + b3[ii];
    tr_i[tid] = st;
  }
  __syncthreads();
  if (tid < NB) {
    float s = 0.0f;
    for (int ii = 0; ii < 32; ++ii) s += tr_i[tid * 32 + ii];
    out[(size_t)BATCH * DDIM + bg + tid] = -s;
  }
}

// ---------- weight pre-conversion f32 -> bf16 into workspace
__global__ void cvt_weights(const float* __restrict__ W1,
                            const float* __restrict__ W2,
                            unsigned short* __restrict__ dst) {
  const size_t n = (size_t)HDIM * HDIM;
  for (size_t idx = (size_t)blockIdx.x * blockDim.x + threadIdx.x; idx < 2 * n;
       idx += (size_t)gridDim.x * blockDim.x) {
    float v = idx < n ? W1[idx] : W2[idx - n];
    dst[idx] = f2bf(v);
  }
}

extern "C" void kernel_launch(void* const* d_in, const int* in_sizes, int n_in,
                              void* d_out, int out_size, void* d_ws,
                              size_t ws_size, hipStream_t stream) {
  (void)in_sizes; (void)n_in; (void)out_size; (void)ws_size;
  const float* t  = (const float*)d_in[0];
  const float* z  = (const float*)d_in[1];
  const float* W0 = (const float*)d_in[2];
  const float* b0 = (const float*)d_in[3];
  const float* W1 = (const float*)d_in[4];
  const float* b1 = (const float*)d_in[5];
  const float* W2 = (const float*)d_in[6];
  const float* b2 = (const float*)d_in[7];
  const float* W3 = (const float*)d_in[8];
  const float* b3 = (const float*)d_in[9];

  unsigned short* wbf = (unsigned short*)d_ws;  // W1bf | W2bf : 4 MB total
  cvt_weights<<<1024, 256, 0, stream>>>(W1, W2, wbf);

  cnf_fused<<<BATCH / NB, 512, LDS_BYTES, stream>>>(
      t, z, W0, b0, wbf, b1, wbf + (size_t)HDIM * HDIM, b2, W3, b3,
      (float*)d_out);
}
